// Attention_13451837571874
// MI455X (gfx1250) — compile-verified, matrix operand order checked by reading, not run on hardware
//
#include <hip/hip_runtime.h>
#include <hip/hip_bf16.h>

typedef __bf16 bf16_t;
typedef __attribute__((ext_vector_type(16))) __bf16 bf16x16;
typedef __attribute__((ext_vector_type(8)))  __bf16 bf16x8;
typedef __attribute__((ext_vector_type(8)))  float  floatx8;

#define B_C    2
#define SEQ_C  2048
#define DIM_C  2048
#define NH_C   16
#define NKV_C  4
#define HD_C   128

// padded LDS strides (in bf16 elements; multiples of 8 keep 16B alignment)
#define AS_STRIDE 40    // 128x32 tile rows padded 32->40
#define KS_STRIDE 136   // 32x128 tile rows padded 128->136

static __device__ inline floatx8 zero8() {
  floatx8 z;
#pragma unroll
  for (int i = 0; i < 8; ++i) z[i] = 0.0f;
  return z;
}

// Build a 16x32 bf16 A/B fragment for v_wmma_f32_16x16x32_bf16.
// Caller passes pointer to (row start + kbase); elements 0..7 come from
// K = kbase+0..7, elements 8..15 from K = kbase+16..23 (per ISA lane layout).
static __device__ inline bf16x16 load_frag(const bf16_t* p) {
  bf16x8 lo = *(const bf16x8*)p;
  bf16x8 hi = *(const bf16x8*)(p + 16);
  bf16x16 f;
#pragma unroll
  for (int i = 0; i < 8; ++i) { f[i] = lo[i]; f[i + 8] = hi[i]; }
  return f;
}

static __device__ inline floatx8 wmma_bf16(bf16x16 a, bf16x16 b, floatx8 c) {
  return __builtin_amdgcn_wmma_f32_16x16x32_bf16(
      /*neg_a=*/false, a, /*neg_b=*/false, b,
      /*c_mod=*/(short)0, c, /*reuse_a=*/false, /*reuse_b=*/false);
}

// ---------------------------------------------------------------- converts
__global__ void k_f32_to_bf16(const float* __restrict__ in,
                              bf16_t* __restrict__ out, long n) {
  long i = (long)blockIdx.x * blockDim.x + threadIdx.x;
  if (i < n) out[i] = (bf16_t)in[i];
}

// in: K x N fp32 (row-major) -> out: N x K bf16 (row-major)
__global__ void k_transpose_bf16(const float* __restrict__ in,
                                 bf16_t* __restrict__ out, int K, int N) {
  long i = (long)blockIdx.x * blockDim.x + threadIdx.x;
  long total = (long)K * N;
  if (i >= total) return;
  int k = (int)(i % K);
  int n = (int)(i / K);
  out[i] = (bf16_t)in[(long)k * N + n];
}

// ---------------------------------------------------------------- GEMM
// C[M,N] fp32 = A[M,K] bf16 @ Bt[N,K]^T bf16. M,N multiples of 128, K of 32.
// LDS-staged: 128x128 block tile, K-step 32. 8 waves; wave (wm,wn) computes a
// 32x64 tile from LDS -> 8 WMMA per K step, 16KB unique global bytes / step.
__global__ __launch_bounds__(256) void k_gemm_bf16(
    const bf16_t* __restrict__ A, const bf16_t* __restrict__ Bt,
    float* __restrict__ C, int M, int N, int K) {
  __shared__ bf16_t As[128 * AS_STRIDE];
  __shared__ bf16_t Bs[128 * AS_STRIDE];
  const int tid = threadIdx.x;
  const int lane = tid & 31;
  const int wave = tid >> 5;
  const int wm = wave & 3, wn = wave >> 2;
  const int hf = lane >> 4, lr = lane & 15;
  const int kb8 = hf * 8;
  const int m_blk = blockIdx.y * 128;
  const int n_blk = blockIdx.x * 128;
  // staging map: thread -> (row 0..127, 16-element segment 0/16)
  const int trow = tid >> 1;
  const int tseg = (tid & 1) * 16;

  floatx8 acc[2][4];
#pragma unroll
  for (int mi = 0; mi < 2; ++mi)
#pragma unroll
    for (int ni = 0; ni < 4; ++ni) acc[mi][ni] = zero8();

  for (int k0 = 0; k0 < K; k0 += 32) {
    const bf16_t* ag = A + (long)(m_blk + trow) * K + k0 + tseg;
    const bf16_t* bg = Bt + (long)(n_blk + trow) * K + k0 + tseg;
    bf16x8 a0 = *(const bf16x8*)ag;
    bf16x8 a1 = *(const bf16x8*)(ag + 8);
    bf16x8 b0 = *(const bf16x8*)bg;
    bf16x8 b1 = *(const bf16x8*)(bg + 8);
    if (k0 + 32 < K) {  // prefetch next K tile (global_prefetch_b8)
      __builtin_prefetch(ag + 32, 0, 1);
      __builtin_prefetch(bg + 32, 0, 1);
    }
    __syncthreads();  // previous iteration's LDS reads complete
    *(bf16x8*)(As + trow * AS_STRIDE + tseg) = a0;
    *(bf16x8*)(As + trow * AS_STRIDE + tseg + 8) = a1;
    *(bf16x8*)(Bs + trow * AS_STRIDE + tseg) = b0;
    *(bf16x8*)(Bs + trow * AS_STRIDE + tseg + 8) = b1;
    __syncthreads();  // tile visible to all waves

    bf16x16 a[2], b[4];
#pragma unroll
    for (int mi = 0; mi < 2; ++mi)
      a[mi] = load_frag(As + (wm * 32 + mi * 16 + lr) * AS_STRIDE + kb8);
#pragma unroll
    for (int ni = 0; ni < 4; ++ni)
      b[ni] = load_frag(Bs + (wn * 64 + ni * 16 + lr) * AS_STRIDE + kb8);
#pragma unroll
    for (int mi = 0; mi < 2; ++mi)
#pragma unroll
      for (int ni = 0; ni < 4; ++ni)
        acc[mi][ni] = wmma_bf16(a[mi], b[ni], acc[mi][ni]);
  }
  // C/D layout: VGPR j -> row j + 8*hf, col lr
#pragma unroll
  for (int mi = 0; mi < 2; ++mi)
#pragma unroll
    for (int ni = 0; ni < 4; ++ni)
#pragma unroll
      for (int j = 0; j < 8; ++j)
        C[(long)(m_blk + wm * 32 + mi * 16 + 8 * hf + j) * N +
          n_blk + wn * 64 + ni * 16 + lr] = acc[mi][ni][j];
}

// ---------------------------------------------------------------- RMSNorm+RoPE
// One block per (b,n,h) row of 128. in: [B*N, H*128] fp32 -> out: [B,H,N,128] bf16
__global__ __launch_bounds__(128) void k_norm_rope(
    const float* __restrict__ in, const float* __restrict__ nw,
    const float* __restrict__ freqs, bf16_t* __restrict__ out,
    int n_heads, int seq) {
  __shared__ float red[128];
  const int idx = blockIdx.x;  // (b*seq + n)*n_heads + h
  const int h = idx % n_heads;
  const int n = (idx / n_heads) % seq;
  const int b = idx / (n_heads * seq);
  const float* row = in + (long)idx * HD_C;
  const int t = threadIdx.x;
  float v = row[t];
  red[t] = v * v;
  __syncthreads();
  for (int s = 64; s > 0; s >>= 1) {
    if (t < s) red[t] += red[t + s];
    __syncthreads();
  }
  float inv = rsqrtf(red[0] * (1.0f / HD_C) + 1e-6f);
  float xn = v * inv * nw[t];
  float xo = row[t ^ 1] * inv * nw[t ^ 1];
  int i = t >> 1;
  float cs = freqs[((long)n * (HD_C / 2) + i) * 2 + 0];
  float sn = freqs[((long)n * (HD_C / 2) + i) * 2 + 1];
  float o = ((t & 1) == 0) ? (xn * cs - xo * sn) : (xo * sn + xn * cs);
  out[((long)(b * n_heads + h) * seq + n) * HD_C + t] = (bf16_t)o;
}

// vf: [B*N, KV*128] fp32 -> vt: [B, KV, 128, N] bf16 (d-major for PV B-frags)
__global__ void k_v_transpose(const float* __restrict__ vf,
                              bf16_t* __restrict__ vt, int seq, int n_kv) {
  long i = (long)blockIdx.x * blockDim.x + threadIdx.x;
  long total = (long)B_C * n_kv * HD_C * seq;
  if (i >= total) return;
  int n = (int)(i % seq);
  long r = i / seq;
  int d = (int)(r % HD_C);
  long r2 = r / HD_C;
  int h = (int)(r2 % n_kv);
  int b = (int)(r2 / n_kv);
  vt[i] = (bf16_t)vf[((long)(b * seq + n) * n_kv + h) * HD_C + d];
}

// ---------------------------------------------------------------- Flash attention
// Q:[B,H,N,128] K:[B,KV,N,128] Vt:[B,KV,128,N] bf16 -> O:[B*N, H*128] bf16
// 8 waves/block handle 128 query rows; all waves share one 32-key K/V tile
// staged in LDS per iteration (K 32x128, V 128x32 d-major, both padded).
__global__ __launch_bounds__(256) void k_flash_attn(
    const bf16_t* __restrict__ Q, const bf16_t* __restrict__ Kc,
    const bf16_t* __restrict__ Vt, bf16_t* __restrict__ O,
    int seq, int n_heads, int n_kv) {
  __shared__ bf16_t Ks[32 * KS_STRIDE];
  __shared__ bf16_t Vs[128 * AS_STRIDE];
  __shared__ bf16_t pbuf[8][16][32];
  const int tid = threadIdx.x;
  const int lane = tid & 31;
  const int wave = tid >> 5;
  const int hf = lane >> 4, lr = lane & 15;
  const int kb8 = hf * 8;
  const int bh = blockIdx.y;
  const int h = bh % n_heads, b = bh / n_heads;
  const int kvh = h / (n_heads / n_kv);
  const int q0 = blockIdx.x * 128 + wave * 16;
  // staging maps
  const int krow = tid >> 3, kseg = (tid & 7) * 16;  // K: 32 rows x 128
  const int vrow = tid >> 1, vseg = (tid & 1) * 16;  // V: 128 rows x 32

  const bf16_t* Qh = Q + (long)bh * seq * HD_C;
  const bf16_t* Kh = Kc + (long)(b * n_kv + kvh) * seq * HD_C;
  const bf16_t* Vh = Vt + (long)(b * n_kv + kvh) * HD_C * seq;

  bf16x16 aq[4];  // Q fragments across head_dim (4 x 32), loaded once
#pragma unroll
  for (int kc = 0; kc < 4; ++kc)
    aq[kc] = load_frag(Qh + (long)(q0 + lr) * HD_C + kc * 32 + kb8);

  floatx8 oacc[8];
#pragma unroll
  for (int dt = 0; dt < 8; ++dt) oacc[dt] = zero8();
  float mrun[8], lrun[8];
#pragma unroll
  for (int j = 0; j < 8; ++j) { mrun[j] = -1e30f; lrun[j] = 0.0f; }
  const float scale = 0.088388347648318447f;  // 1/sqrt(128)

  for (int kblk = 0; kblk < seq; kblk += 32) {
    // ---- stage K/V tile (coalesced 32B per thread per tile)
    const bf16_t* kg = Kh + (long)(kblk + krow) * HD_C + kseg;
    const bf16_t* vg = Vh + (long)vrow * seq + kblk + vseg;
    bf16x8 k0v = *(const bf16x8*)kg;
    bf16x8 k1v = *(const bf16x8*)(kg + 8);
    bf16x8 v0v = *(const bf16x8*)vg;
    bf16x8 v1v = *(const bf16x8*)(vg + 8);
    __syncthreads();  // previous iteration's LDS reads complete
    *(bf16x8*)(Ks + krow * KS_STRIDE + kseg) = k0v;
    *(bf16x8*)(Ks + krow * KS_STRIDE + kseg + 8) = k1v;
    *(bf16x8*)(Vs + vrow * AS_STRIDE + vseg) = v0v;
    *(bf16x8*)(Vs + vrow * AS_STRIDE + vseg + 8) = v1v;
    __syncthreads();

    // ---- scores: two 16x16 tiles over 32 keys
    floatx8 s0 = zero8(), s1 = zero8();
#pragma unroll
    for (int kc = 0; kc < 4; ++kc) {
      bf16x16 b0 = load_frag(Ks + lr * KS_STRIDE + kc * 32 + kb8);
      bf16x16 b1 = load_frag(Ks + (16 + lr) * KS_STRIDE + kc * 32 + kb8);
      s0 = wmma_bf16(aq[kc], b0, s0);
      s1 = wmma_bf16(aq[kc], b1, s1);
    }
    // ---- online softmax. Element j is row (j + 8*hf); each row spans the
    // 16 lanes of one half, so xor-reduce with masks 1,2,4,8 only.
#pragma unroll
    for (int j = 0; j < 8; ++j) {
      float x0 = s0[j] * scale;
      float x1 = s1[j] * scale;
      float mx = fmaxf(x0, x1);
#pragma unroll
      for (int off = 1; off < 16; off <<= 1)
        mx = fmaxf(mx, __shfl_xor(mx, off, 32));
      float mnew = fmaxf(mrun[j], mx);
      float alpha = __expf(mrun[j] - mnew);
      float p0 = __expf(x0 - mnew);
      float p1 = __expf(x1 - mnew);
      float rs = p0 + p1;
#pragma unroll
      for (int off = 1; off < 16; off <<= 1)
        rs += __shfl_xor(rs, off, 32);
      lrun[j] = lrun[j] * alpha + rs;
      mrun[j] = mnew;
#pragma unroll
      for (int dt = 0; dt < 8; ++dt) oacc[dt][j] *= alpha;
      pbuf[wave][j + 8 * hf][lr] = (bf16_t)p0;
      pbuf[wave][j + 8 * hf][16 + lr] = (bf16_t)p1;
    }
    __syncthreads();  // pbuf C-layout -> A-frag layout redistribution
    bf16x16 ap = load_frag(&pbuf[wave][lr][kb8]);  // P as A-frag (16x32)
#pragma unroll
    for (int dt = 0; dt < 8; ++dt) {
      bf16x16 bv = load_frag(Vs + (dt * 16 + lr) * AS_STRIDE + kb8);
      oacc[dt] = wmma_bf16(ap, bv, oacc[dt]);
    }
  }
#pragma unroll
  for (int j = 0; j < 8; ++j) {
    float invl = 1.0f / lrun[j];
    int row = q0 + j + 8 * hf;
    long base = (long)(b * seq + row) * (n_heads * HD_C) + h * HD_C;
#pragma unroll
    for (int dt = 0; dt < 8; ++dt)
      O[base + dt * 16 + lr] = (bf16_t)(oacc[dt][j] * invl);
  }
}

// ---------------------------------------------------------------- launcher
extern "C" void kernel_launch(void* const* d_in, const int* in_sizes, int n_in,
                              void* d_out, int out_size, void* d_ws,
                              size_t ws_size, hipStream_t stream) {
  (void)in_sizes; (void)n_in; (void)out_size; (void)ws_size;
  const float* x     = (const float*)d_in[0];
  const float* wq    = (const float*)d_in[1];
  const float* wk    = (const float*)d_in[2];
  const float* wv    = (const float*)d_in[3];
  const float* wpr   = (const float*)d_in[4];
  const float* qnw   = (const float*)d_in[5];
  const float* knw   = (const float*)d_in[6];
  const float* freqs = (const float*)d_in[7];
  float* out = (float*)d_out;

  const int M = B_C * SEQ_C;          // 4096 rows
  const int KVD = NKV_C * HD_C;       // 512

  char* ws = (char*)d_ws;
  size_t off = 0;
  auto take = [&](size_t bytes) -> void* {
    void* p = ws + off;
    off += (bytes + 255) & ~(size_t)255;
    return p;
  };
  bf16_t* xb  = (bf16_t*)take((size_t)M * DIM_C * 2);
  bf16_t* wqT = (bf16_t*)take((size_t)DIM_C * DIM_C * 2);
  bf16_t* wkT = (bf16_t*)take((size_t)KVD * DIM_C * 2);
  bf16_t* wvT = (bf16_t*)take((size_t)KVD * DIM_C * 2);
  bf16_t* wpT = (bf16_t*)take((size_t)DIM_C * DIM_C * 2);
  float*  qf  = (float*)take((size_t)M * DIM_C * 4);
  float*  kf  = (float*)take((size_t)M * KVD * 4);
  float*  vf  = (float*)take((size_t)M * KVD * 4);
  bf16_t* qbf = (bf16_t*)take((size_t)M * DIM_C * 2);
  bf16_t* kbf = (bf16_t*)take((size_t)M * KVD * 2);
  bf16_t* vtb = (bf16_t*)take((size_t)M * KVD * 2);
  bf16_t* aob = (bf16_t*)take((size_t)M * DIM_C * 2);

  const int T = 256;
  long nx = (long)M * DIM_C;
  k_f32_to_bf16<<<(unsigned)((nx + T - 1) / T), T, 0, stream>>>(x, xb, nx);

  long nqq = (long)DIM_C * DIM_C;
  long nkk = (long)DIM_C * KVD;
  k_transpose_bf16<<<(unsigned)((nqq + T - 1) / T), T, 0, stream>>>(wq, wqT, DIM_C, DIM_C);
  k_transpose_bf16<<<(unsigned)((nkk + T - 1) / T), T, 0, stream>>>(wk, wkT, DIM_C, KVD);
  k_transpose_bf16<<<(unsigned)((nkk + T - 1) / T), T, 0, stream>>>(wv, wvT, DIM_C, KVD);
  k_transpose_bf16<<<(unsigned)((nqq + T - 1) / T), T, 0, stream>>>(wpr, wpT, DIM_C, DIM_C);

  dim3 gq(DIM_C / 128, M / 128);
  dim3 gkv(KVD / 128, M / 128);
  k_gemm_bf16<<<gq, 256, 0, stream>>>(xb, wqT, qf, M, DIM_C, DIM_C);
  k_gemm_bf16<<<gkv, 256, 0, stream>>>(xb, wkT, kf, M, KVD, DIM_C);
  k_gemm_bf16<<<gkv, 256, 0, stream>>>(xb, wvT, vf, M, KVD, DIM_C);

  k_norm_rope<<<M * NH_C, 128, 0, stream>>>(qf, qnw, freqs, qbf, NH_C, SEQ_C);
  k_norm_rope<<<M * NKV_C, 128, 0, stream>>>(kf, knw, freqs, kbf, NKV_C, SEQ_C);

  long nv = (long)B_C * NKV_C * HD_C * SEQ_C;
  k_v_transpose<<<(unsigned)((nv + T - 1) / T), T, 0, stream>>>(vf, vtb, SEQ_C, NKV_C);

  dim3 ga(SEQ_C / 128, B_C * NH_C);
  k_flash_attn<<<ga, 256, 0, stream>>>(qbf, kbf, vtb, aob, SEQ_C, NH_C, NKV_C);

  k_gemm_bf16<<<gq, 256, 0, stream>>>(aob, wpT, out, M, DIM_C, DIM_C);
}
